// ProbabilisticRegionCollapse_62715112456627
// MI455X (gfx1250) — matrix operand order repaired and checked by reading, not compiled, not run on hardware
//
#include <hip/hip_runtime.h>

#define DI __device__ __forceinline__

typedef __attribute__((ext_vector_type(16))) __bf16 v16bf;
typedef __attribute__((ext_vector_type(8)))  float  v8f;
typedef __attribute__((ext_vector_type(4)))  unsigned int u32x4;

union FragU {
  v16bf bf;
  u32x4 u[2];
};

// ---------- constants ----------
static constexpr int B_  = 4;
static constexpr int S_  = 2048;
static constexpr int D_  = 1024;
static constexpr int N_  = 16384;
static constexpr int K_  = 8;
static constexpr int BS_ = B_ * S_;        // 8192 tokens
static constexpr int KD_ = K_ * D_;        // 8192 reduction for mu/sigma

// ---------- helpers ----------
DI unsigned short f2bf(float f) {
  unsigned int u = __float_as_uint(f);
  u += 0x7FFFu + ((u >> 16) & 1u);        // round-to-nearest-even
  return (unsigned short)(u >> 16);
}

DI float sigmoidf_(float x) { return 1.0f / (1.0f + __expf(-x)); }

// CDNA5 async global->LDS copy (ASYNCcnt-tracked), 16 bytes per lane.
DI void async_copy_b128(unsigned lds_addr, const void* gaddr) {
  asm volatile("global_load_async_to_lds_b128 %0, %1, off"
               :: "v"(lds_addr), "v"(gaddr) : "memory");
}
DI void wait_async0() { asm volatile("s_wait_asynccnt 0x0" ::: "memory"); }
DI unsigned lds_off(const void* p) { return (unsigned)(uintptr_t)p; }

// A-matrix fragment (16x32 bf16): lane L(0..15) holds row M=L, K = kk+[0..7] and kk+[16..23];
// lanes 16..31 hold same rows with K = kk+[8..15] and kk+[24..31].
DI v16bf load_a_frag(const unsigned short* row, int kk, int half) {
  FragU f;
  f.u[0] = *(const u32x4*)(row + kk + 8 * half);
  f.u[1] = *(const u32x4*)(row + kk + 16 + 8 * half);
  return f.bf;
}

// B-matrix fragment (32x16 bf16): lane n(0..15) holds col N=n, K = kk+[0..7] and kk+[16..23];
// lanes 16..31 hold K = kk+[8..15] and kk+[24..31]. "row" points at W[n][*] (row-major weight row).
DI v16bf load_b_frag(const unsigned short* row, int kk, int half) {
  FragU f;
  const int o = kk + 8 * half;
  f.u[0] = *(const u32x4*)(row + o);
  f.u[1] = *(const u32x4*)(row + o + 16);
  return f.bf;
}

DI v8f wmma_bf16(v16bf a, v16bf b, v8f c) {
  return __builtin_amdgcn_wmma_f32_16x16x32_bf16(false, a, false, b, (short)0, c,
                                                 false, false);
}

// ---------- elementwise f32 -> bf16 ----------
__global__ __launch_bounds__(256) void cvt_bf16_kernel(const float* __restrict__ src,
                                                       unsigned short* __restrict__ dst, int n) {
  int i = blockIdx.x * blockDim.x + threadIdx.x;
  const int stride = gridDim.x * blockDim.x;
  for (; i < n; i += stride) dst[i] = f2bf(src[i]);
}

// ---------- row |x|^2 over D=1024, one wave32 per row ----------
__global__ __launch_bounds__(256) void rownorm_kernel(const float* __restrict__ src,
                                                      float* __restrict__ out, int rows) {
  const int wave = threadIdx.x >> 5;
  const int lane = threadIdx.x & 31;
  const int row = blockIdx.x * 8 + wave;
  if (row >= rows) return;
  const float* p = src + (size_t)row * D_;
  float s = 0.0f;
  #pragma unroll 8
  for (int i = lane; i < D_; i += 32) { float v = p[i]; s += v * v; }
  #pragma unroll
  for (int m = 16; m > 0; m >>= 1) s += __shfl_xor(s, m, 32);
  if (lane == 0) out[row] = s;
}

// ---------- fused distance GEMM + per-query top-8 (selection on clamped d^2) ----------
// block: 256 threads (8 waves), 16 queries. A = concepts (M), B = queries (N).
// Each wave sweeps 2 concept tiles per chunk (2 accumulators share one LDS B fragment).
__global__ __launch_bounds__(256) void topk_kernel(
    const unsigned short* __restrict__ q_bf,
    const unsigned short* __restrict__ cb_bf,
    const float* __restrict__ qn,
    const float* __restrict__ cn,
    float* __restrict__ out_dist,
    int* __restrict__ out_idx) {
  __shared__ unsigned short s_q[16 * D_];     // 32 KB
  __shared__ float s_md[16 * 128];            // 8 KB
  __shared__ int   s_mi[16 * 128];            // 8 KB

  const int tid  = threadIdx.x;
  const int wave = tid >> 5;
  const int lane = tid & 31;
  const int half = lane >> 4;
  const int l15  = lane & 15;
  const int tbase = blockIdx.x * 16;

  // async-stage 16 query rows (bf16) into LDS: 2048 x 16B chunks
  {
    const unsigned lb = lds_off(&s_q[0]);
    const unsigned short* gq = q_bf + (size_t)tbase * D_;
    #pragma unroll
    for (int i = tid; i < 2048; i += 256)
      async_copy_b128(lb + i * 16, (const void*)(gq + i * 8));
    wait_async0();
  }
  __syncthreads();

  const float my_qn = qn[tbase + l15];
  const unsigned short* qrow = s_q + l15 * D_;

  float bd[8]; int bi[8];
  #pragma unroll
  for (int j = 0; j < 8; ++j) { bd[j] = 3.0e38f; bi[j] = 0; }

  for (int chunk = 0; chunk < N_; chunk += 256) {
    const int cbase0 = chunk + wave * 32;
    const int cbase1 = cbase0 + 16;
    const unsigned short* arow0 = cb_bf + (size_t)(cbase0 + l15) * D_;
    const unsigned short* arow1 = cb_bf + (size_t)(cbase1 + l15) * D_;
    v8f c0 = {}, c1 = {};
    #pragma unroll 4
    for (int kk = 0; kk < D_; kk += 32) {
      v16bf b  = load_b_frag(qrow, kk, half);
      v16bf a0 = load_a_frag(arow0, kk, half);
      v16bf a1 = load_a_frag(arow1, kk, half);
      c0 = wmma_bf16(a0, b, c0);
      c1 = wmma_bf16(a1, b, c1);
    }
    #pragma unroll
    for (int r = 0; r < 8; ++r) {
      // selection metric: clamped squared distance (sqrt deferred to merge)
      const int ci0 = cbase0 + 8 * half + r;
      const float d20 = fmaxf(my_qn + cn[ci0] - 2.0f * c0[r], 0.0f);
      if (d20 < bd[7]) {
        bd[7] = d20; bi[7] = ci0;
        #pragma unroll
        for (int p = 7; p > 0; --p)
          if (bd[p] < bd[p - 1]) {
            float td = bd[p]; bd[p] = bd[p - 1]; bd[p - 1] = td;
            int ti = bi[p]; bi[p] = bi[p - 1]; bi[p - 1] = ti;
          }
      }
      const int ci1 = cbase1 + 8 * half + r;
      const float d21 = fmaxf(my_qn + cn[ci1] - 2.0f * c1[r], 0.0f);
      if (d21 < bd[7]) {
        bd[7] = d21; bi[7] = ci1;
        #pragma unroll
        for (int p = 7; p > 0; --p)
          if (bd[p] < bd[p - 1]) {
            float td = bd[p]; bd[p] = bd[p - 1]; bd[p - 1] = td;
            int ti = bi[p]; bi[p] = bi[p - 1]; bi[p - 1] = ti;
          }
      }
    }
  }

  // merge: 8 waves x 2 half-lanes x 8 entries = 128 candidates per query
  const int slot = (wave * 2 + half) * 8;
  #pragma unroll
  for (int j = 0; j < 8; ++j) {
    s_md[l15 * 128 + slot + j] = bd[j];
    s_mi[l15 * 128 + slot + j] = bi[j];
  }
  __syncthreads();

  if (tid < 16) {
    float* md = s_md + tid * 128;
    int*   mi = s_mi + tid * 128;
    for (int s = 0; s < 8; ++s) {
      float best = md[0]; int be = 0;
      for (int e = 1; e < 128; ++e)
        if (md[e] < best) { best = md[e]; be = e; }
      out_dist[(size_t)(tbase + tid) * K_ + s] = __builtin_amdgcn_sqrtf(best);
      out_idx [(size_t)(tbase + tid) * K_ + s] = mi[be];
      md[be] = 3.0e38f;
    }
  }
}

// ---------- mu_lin / sig_lin GEMM with gathered, async double-buffered A tile ----------
// grid: (BS/16, D/128); wave w covers dims [by*128 + w*16, +16). A tile (16 tokens x 1024)
// for neighbor segment s is staged once per block into LDS while segment s-1 computes.
__global__ __launch_bounds__(256) void musig_gemm_kernel(
    const unsigned short* __restrict__ cb_bf,
    const int* __restrict__ topk_idx,
    const unsigned short* __restrict__ wmu_bf,
    const unsigned short* __restrict__ wsig_bf,
    float* __restrict__ mu_lin,
    float* __restrict__ sig_lin) {
  __shared__ unsigned short s_a[2][16 * D_];  // 2 x 32 KB double buffer

  const int tid  = threadIdx.x;
  const int wave = tid >> 5;
  const int lane = tid & 31;
  const int half = lane >> 4;
  const int l15  = lane & 15;
  const int tbase = blockIdx.x * 16;
  const int dbase = blockIdx.y * 128 + wave * 16;

  const unsigned short* bmu_row  = wmu_bf  + (size_t)(dbase + l15) * KD_;
  const unsigned short* bsig_row = wsig_bf + (size_t)(dbase + l15) * KD_;

  // stage segment 0
  {
    const unsigned lb = lds_off(&s_a[0][0]);
    #pragma unroll
    for (int i = tid; i < 2048; i += 256) {
      const int row = i >> 7;                 // 128 x16B chunks per row
      const int col = (i & 127) * 8;
      const int cidx = topk_idx[(tbase + row) * K_ + 0];
      async_copy_b128(lb + i * 16, (const void*)(cb_bf + (size_t)cidx * D_ + col));
    }
    wait_async0();
  }
  __syncthreads();

  v8f cmu = {}, csig = {};
  for (int seg = 0; seg < K_; ++seg) {
    // prefetch next neighbor segment into the other buffer
    if (seg + 1 < K_) {
      const unsigned lb = lds_off(&s_a[(seg + 1) & 1][0]);
      #pragma unroll
      for (int i = tid; i < 2048; i += 256) {
        const int row = i >> 7;
        const int col = (i & 127) * 8;
        const int cidx = topk_idx[(tbase + row) * K_ + seg + 1];
        async_copy_b128(lb + i * 16, (const void*)(cb_bf + (size_t)cidx * D_ + col));
      }
    }
    const unsigned short* arow = &s_a[seg & 1][0] + l15 * D_;
    const int kbase = seg * D_;
    #pragma unroll 4
    for (int kk = 0; kk < D_; kk += 32) {
      v16bf a  = load_a_frag(arow, kk, half);
      v16bf b0 = load_b_frag(bmu_row, kbase + kk, half);
      v16bf b1 = load_b_frag(bsig_row, kbase + kk, half);
      cmu  = wmma_bf16(a, b0, cmu);
      csig = wmma_bf16(a, b1, csig);
    }
    if (seg + 1 < K_) { wait_async0(); }
    __syncthreads();
  }

  const int dim = dbase + l15;
  #pragma unroll
  for (int r = 0; r < 8; ++r) {
    const int t = tbase + 8 * half + r;
    mu_lin [(size_t)t * D_ + dim] = cmu[r];
    sig_lin[(size_t)t * D_ + dim] = csig[r];
  }
}

// ---------- softmax(K=8) + weighted-mu gather + reparameterized sample ----------
__global__ __launch_bounds__(256) void sample_kernel(
    const float* __restrict__ mu_lin,
    const float* __restrict__ sig_lin,
    const float* __restrict__ bmu,
    const float* __restrict__ bsig,
    const float* __restrict__ cbank,
    const int* __restrict__ topk_idx,
    const float* __restrict__ topk_dist,
    const float* __restrict__ eps,
    float* __restrict__ samples,
    unsigned short* __restrict__ samples_bf) {
  const int t = blockIdx.x;
  float w[8]; int id[8];
  float mx = -3.0e38f;
  #pragma unroll
  for (int k = 0; k < 8; ++k) {
    w[k]  = -topk_dist[t * K_ + k];
    id[k] = topk_idx[t * K_ + k];
    mx = fmaxf(mx, w[k]);
  }
  float sum = 0.0f;
  #pragma unroll
  for (int k = 0; k < 8; ++k) { w[k] = __expf(w[k] - mx); sum += w[k]; }
  const float inv = 1.0f / sum;
  #pragma unroll
  for (int k = 0; k < 8; ++k) w[k] *= inv;

  for (int d = threadIdx.x; d < D_; d += 256) {
    float wm = 0.0f;
    #pragma unroll
    for (int k = 0; k < 8; ++k) wm += w[k] * cbank[(size_t)id[k] * D_ + d];
    const size_t o = (size_t)t * D_ + d;
    const float mu = 0.5f * (mu_lin[o] + bmu[d]) + 0.5f * wm;
    const float sg = sigmoidf_(sig_lin[o] + bsig[d]) * 0.5f;   // SIGMA_MAX * SIGMA_CTRL
    const float s = mu + sg * eps[o];
    samples[o] = s;
    samples_bf[o] = f2bf(s);
  }
}

// ---------- gate layer 1: h1 = relu([h|s] @ Wg1.T + bg1) ----------
// grid: (BS/16, D/256); wave covers 2 dim-tiles (32 dims); A tile (16 x 2048) async-staged in LDS.
__global__ __launch_bounds__(256) void gate1_gemm_kernel(
    const unsigned short* __restrict__ q_bf,
    const unsigned short* __restrict__ samp_bf,
    const unsigned short* __restrict__ wg1_bf,
    const float* __restrict__ bg1,
    unsigned short* __restrict__ h1_bf) {
  __shared__ unsigned short s_a[16 * 2 * D_];   // 64 KB

  const int tid  = threadIdx.x;
  const int wave = tid >> 5;
  const int lane = tid & 31;
  const int half = lane >> 4;
  const int l15  = lane & 15;
  const int tbase = blockIdx.x * 16;
  const int dbase = blockIdx.y * 256 + wave * 32;

  {
    const unsigned lb = lds_off(&s_a[0]);
    #pragma unroll
    for (int i = tid; i < 4096; i += 256) {       // 16 rows x 256 x16B chunks
      const int row = i >> 8;
      const int cw = i & 255;
      const unsigned short* src = (cw < 128)
          ? (q_bf   + (size_t)(tbase + row) * D_ + cw * 8)
          : (samp_bf + (size_t)(tbase + row) * D_ + (cw - 128) * 8);
      async_copy_b128(lb + i * 16, (const void*)src);
    }
    wait_async0();
  }
  __syncthreads();

  const unsigned short* arow = &s_a[0] + l15 * (2 * D_);
  const unsigned short* brow0 = wg1_bf + (size_t)(dbase + l15) * (2 * D_);
  const unsigned short* brow1 = wg1_bf + (size_t)(dbase + 16 + l15) * (2 * D_);

  v8f c0 = {}, c1 = {};
  #pragma unroll 4
  for (int kk = 0; kk < 2 * D_; kk += 32) {
    v16bf a  = load_a_frag(arow, kk, half);
    v16bf b0 = load_b_frag(brow0, kk, half);
    v16bf b1 = load_b_frag(brow1, kk, half);
    c0 = wmma_bf16(a, b0, c0);
    c1 = wmma_bf16(a, b1, c1);
  }
  const int dim0 = dbase + l15, dim1 = dbase + 16 + l15;
  const float bias0 = bg1[dim0], bias1 = bg1[dim1];
  #pragma unroll
  for (int r = 0; r < 8; ++r) {
    const int t = tbase + 8 * half + r;
    h1_bf[(size_t)t * D_ + dim0] = f2bf(fmaxf(c0[r] + bias0, 0.0f));
    h1_bf[(size_t)t * D_ + dim1] = f2bf(fmaxf(c1[r] + bias1, 0.0f));
  }
}

// ---------- gate layer 2 + final blend ----------
__global__ __launch_bounds__(256) void gate2_final_kernel(
    const unsigned short* __restrict__ h1_bf,
    const unsigned short* __restrict__ wg2_bf,
    const float* __restrict__ bg2,
    const float* __restrict__ gamma,
    const unsigned char* __restrict__ trig,
    const float* __restrict__ hidden,
    const float* __restrict__ samples,
    float* __restrict__ out) {
  __shared__ unsigned short s_a[16 * D_];       // 32 KB

  const int tid  = threadIdx.x;
  const int wave = tid >> 5;
  const int lane = tid & 31;
  const int half = lane >> 4;
  const int l15  = lane & 15;
  const int tbase = blockIdx.x * 16;
  const int dbase = blockIdx.y * 256 + wave * 32;

  {
    const unsigned lb = lds_off(&s_a[0]);
    const unsigned short* g = h1_bf + (size_t)tbase * D_;
    #pragma unroll
    for (int i = tid; i < 2048; i += 256)
      async_copy_b128(lb + i * 16, (const void*)(g + i * 8));
    wait_async0();
  }
  __syncthreads();

  const unsigned short* arow = &s_a[0] + l15 * D_;
  const unsigned short* brow0 = wg2_bf + (size_t)(dbase + l15) * D_;
  const unsigned short* brow1 = wg2_bf + (size_t)(dbase + 16 + l15) * D_;

  v8f c0 = {}, c1 = {};
  #pragma unroll 4
  for (int kk = 0; kk < D_; kk += 32) {
    v16bf a  = load_a_frag(arow, kk, half);
    v16bf b0 = load_b_frag(brow0, kk, half);
    v16bf b1 = load_b_frag(brow1, kk, half);
    c0 = wmma_bf16(a, b0, c0);
    c1 = wmma_bf16(a, b1, c1);
  }
  const float g0 = gamma[0];              // * GAMMA_CTRL (1.0)
  const int dim0 = dbase + l15, dim1 = dbase + 16 + l15;
  const float bias0 = bg2[dim0], bias1 = bg2[dim1];
  #pragma unroll
  for (int r = 0; r < 8; ++r) {
    const int t = tbase + 8 * half + r;
    const float m = trig[t] ? 1.0f : 0.0f;
    {
      const float gate = sigmoidf_(c0[r] + bias0) * g0 * m;
      const size_t o = (size_t)t * D_ + dim0;
      const float h = hidden[o], s = samples[o];
      out[o] = h + gate * (s - h);
    }
    {
      const float gate = sigmoidf_(c1[r] + bias1) * g0 * m;
      const size_t o = (size_t)t * D_ + dim1;
      const float h = hidden[o], s = samples[o];
      out[o] = h + gate * (s - h);
    }
  }
}

// ---------- launch ----------
extern "C" void kernel_launch(void* const* d_in, const int* in_sizes, int n_in,
                              void* d_out, int out_size, void* d_ws, size_t ws_size,
                              hipStream_t stream) {
  (void)in_sizes; (void)n_in; (void)out_size; (void)ws_size;
  const float*         hidden = (const float*)d_in[0];
  const unsigned char* trig   = (const unsigned char*)d_in[1];   // bool mask
  const float*         cbank  = (const float*)d_in[2];
  const float*         Wmu    = (const float*)d_in[3];
  const float*         bmu    = (const float*)d_in[4];
  const float*         Wsig   = (const float*)d_in[5];
  const float*         bsig   = (const float*)d_in[6];
  const float*         Wg1    = (const float*)d_in[7];
  const float*         bg1    = (const float*)d_in[8];
  const float*         Wg2    = (const float*)d_in[9];
  const float*         bg2    = (const float*)d_in[10];
  const float*         gamma  = (const float*)d_in[11];
  const float*         eps    = (const float*)d_in[12];
  float* out = (float*)d_out;

  char* ws = (char*)d_ws;
  size_t off = 0;
  auto carve = [&](size_t bytes) { char* p = ws + off; off += (bytes + 255) & ~(size_t)255; return p; };

  unsigned short* cb_bf   = (unsigned short*)carve((size_t)N_ * D_ * 2);   // 32 MB
  unsigned short* q_bf    = (unsigned short*)carve((size_t)BS_ * D_ * 2);  // 16 MB
  unsigned short* wmu_bf  = (unsigned short*)carve((size_t)D_ * KD_ * 2);  // 16 MB
  unsigned short* wsig_bf = (unsigned short*)carve((size_t)D_ * KD_ * 2);  // 16 MB
  unsigned short* wg1_bf  = (unsigned short*)carve((size_t)D_ * 2 * D_ * 2); // 4 MB
  unsigned short* wg2_bf  = (unsigned short*)carve((size_t)D_ * D_ * 2);   // 2 MB
  float* cn       = (float*)carve((size_t)N_ * 4);
  float* qn       = (float*)carve((size_t)BS_ * 4);
  int*   tk_idx   = (int*)carve((size_t)BS_ * K_ * 4);
  float* tk_dist  = (float*)carve((size_t)BS_ * K_ * 4);
  float* mu_lin   = (float*)carve((size_t)BS_ * D_ * 4);   // 32 MB
  float* sig_lin  = (float*)carve((size_t)BS_ * D_ * 4);   // 32 MB
  float* samples  = (float*)carve((size_t)BS_ * D_ * 4);   // 32 MB
  unsigned short* samp_bf = (unsigned short*)carve((size_t)BS_ * D_ * 2);  // 16 MB
  unsigned short* h1_bf   = (unsigned short*)carve((size_t)BS_ * D_ * 2);  // 16 MB

  // 1) precision conversion + norms
  cvt_bf16_kernel<<<2048, 256, 0, stream>>>(cbank, cb_bf, N_ * D_);
  cvt_bf16_kernel<<<1024, 256, 0, stream>>>(hidden, q_bf, BS_ * D_);
  cvt_bf16_kernel<<<1024, 256, 0, stream>>>(Wmu, wmu_bf, D_ * KD_);
  cvt_bf16_kernel<<<1024, 256, 0, stream>>>(Wsig, wsig_bf, D_ * KD_);
  cvt_bf16_kernel<<<512, 256, 0, stream>>>(Wg1, wg1_bf, D_ * 2 * D_);
  cvt_bf16_kernel<<<512, 256, 0, stream>>>(Wg2, wg2_bf, D_ * D_);
  rownorm_kernel<<<N_ / 8, 256, 0, stream>>>(cbank, cn, N_);
  rownorm_kernel<<<BS_ / 8, 256, 0, stream>>>(hidden, qn, BS_);

  // 2) fused distance GEMM + top-8
  topk_kernel<<<BS_ / 16, 256, 0, stream>>>(q_bf, cb_bf, qn, cn, tk_dist, tk_idx);

  // 3) mu/sigma GEMMs with gathered, double-buffered A
  dim3 musig_grid(BS_ / 16, D_ / 128);
  musig_gemm_kernel<<<musig_grid, 256, 0, stream>>>(cb_bf, tk_idx, wmu_bf, wsig_bf,
                                                    mu_lin, sig_lin);

  // 4) softmax + weighted-mu + sample
  sample_kernel<<<BS_, 256, 0, stream>>>(mu_lin, sig_lin, bmu, bsig, cbank,
                                         tk_idx, tk_dist, eps, samples, samp_bf);

  // 5) gate MLP + final blend
  dim3 gate_grid(BS_ / 16, D_ / 256);
  gate1_gemm_kernel<<<gate_grid, 256, 0, stream>>>(q_bf, samp_bf, wg1_bf, bg1, h1_bf);
  gate2_final_kernel<<<gate_grid, 256, 0, stream>>>(h1_bf, wg2_bf, bg2, gamma, trig,
                                                    hidden, samples, out);
}
